// GraphRulesSingleRelationship_65120294142436
// MI455X (gfx1250) — matrix operand order repaired and checked by reading, not compiled, not run on hardware
//
#include <hip/hip_runtime.h>
#include <hip/hip_bf16.h>

typedef __attribute__((ext_vector_type(16))) _Float16 v16h;
typedef __attribute__((ext_vector_type(8)))  float    v8f;
typedef __attribute__((ext_vector_type(4)))  unsigned uv4;

namespace {

constexpr int FIN  = 8;
constexpr int Hh   = 64;      // H
constexpr int H2   = 128;     // 2H
constexpr int SS   = 4;       // samples
constexpr int IMGD = 512;
constexpr int C1   = 32;      // conv1 out channels
constexpr int FM   = 128;     // fm spatial (after two stride-2 convs)
constexpr int TE   = 16;      // edges per wave (WMMA M)
constexpr int WV   = 4;       // waves per block

// ---------------- utility kernels ----------------
__global__ void k_zero(float* p, int n) {
  int i = blockIdx.x * blockDim.x + threadIdx.x;
  if (i < n) p[i] = 0.f;
}

__global__ void k_deg(const int* dst, float* deg, int E) {
  int i = blockIdx.x * blockDim.x + threadIdx.x;
  if (i < E) atomicAdd(&deg[dst[i]], 1.f);
}

__global__ void k_norm(float* nrm, int n) {
  int i = blockIdx.x * blockDim.x + threadIdx.x;
  if (i < n) nrm[i] = rsqrtf(nrm[i] + 1.f);
}

// hn = (X @ W) * norm[i]
__global__ void k_mm_scale(const float* X, const float* W, const float* nrm,
                           float* Y, int n, int K, int F) {
  int idx = blockIdx.x * blockDim.x + threadIdx.x;
  if (idx >= n * F) return;
  int i = idx / F, f = idx % F;
  float acc = 0.f;
  for (int k = 0; k < K; ++k) acc += X[i * K + k] * W[k * F + f];
  Y[idx] = acc * nrm[i];
}

__global__ void k_scatter(const int* src, const int* dst, const float* hn,
                          float* agg, int E, int F) {
  int idx = blockIdx.x * blockDim.x + threadIdx.x;
  if (idx >= E * F) return;
  int e = idx / F, f = idx % F;
  atomicAdd(&agg[dst[e] * F + f], hn[src[e] * F + f]);
}

__global__ void k_finish(const float* agg, const float* hn, const float* nrm,
                         const float* b, float* out, _Float16* out16, int n, int F) {
  int idx = blockIdx.x * blockDim.x + threadIdx.x;
  if (idx >= n * F) return;
  int i = idx / F, f = idx % F;
  float v = (agg[idx] + hn[idx]) * nrm[i] + b[f];
  v = fmaxf(v, 0.f);
  out[idx] = v;
  if (out16) out16[idx] = (_Float16)v;
}

// ---------------- conv kernels (TF-SAME, stride 2, 3x3) ----------------
__global__ void k_conv1(const float* img, const float* k1, const float* bk1, float* out) {
  int idx = blockIdx.x * blockDim.x + threadIdx.x;      // 32*256*256
  if (idx >= C1 * 256 * 256) return;
  int ox = idx & 255, oy = (idx >> 8) & 255, o = idx >> 16;
  float acc = 0.f;
  for (int i = 0; i < 3; ++i)
    for (int ky = 0; ky < 3; ++ky) {
      int iy = oy * 2 + ky;
      if (iy >= IMGD) continue;
      for (int kx = 0; kx < 3; ++kx) {
        int ix = ox * 2 + kx;
        if (ix >= IMGD) continue;
        acc += img[(i * IMGD + iy) * IMGD + ix] * k1[((o * 3 + i) * 3 + ky) * 3 + kx];
      }
    }
  out[idx] = fmaxf(acc + bk1[o], 0.f);
}

__global__ void k_conv2(const float* in, const float* k2, const float* bk2, float* out) {
  int idx = blockIdx.x * blockDim.x + threadIdx.x;      // 64*128*128
  if (idx >= Hh * FM * FM) return;
  int ox = idx & 127, oy = (idx >> 7) & 127, o = idx >> 14;
  float acc = 0.f;
  for (int i = 0; i < C1; ++i)
    for (int ky = 0; ky < 3; ++ky) {
      int iy = oy * 2 + ky;
      if (iy >= 256) continue;
      for (int kx = 0; kx < 3; ++kx) {
        int ix = ox * 2 + kx;
        if (ix >= 256) continue;
        acc += in[(i * 256 + iy) * 256 + ix] * k2[((o * C1 + i) * 3 + ky) * 3 + kx];
      }
    }
  out[idx] = fmaxf(acc + bk2[o], 0.f);
}

// ---------------- bilinear box sampling -> f16 feature table ----------------
__global__ void k_sample(const float* fm, const float* pos, const float* cw,
                         _Float16* imgf, int n) {
  int idx = blockIdx.x * blockDim.x + threadIdx.x;      // n * 4 * 64
  if (idx >= n * SS * Hh) return;
  int c = idx & 63, s = (idx >> 6) & 3, nd = idx >> 8;
  const float offx[4] = {-0.25f, 0.25f, -0.25f, 0.25f};
  const float offy[4] = {-0.25f, -0.25f, 0.25f, 0.25f};
  float px = pos[nd * 2 + 0] + offx[s] * cw[nd * 2 + 0];
  float py = pos[nd * 2 + 1] + offy[s] * cw[nd * 2 + 1];
  px = fminf(fmaxf(px, 0.f), 1.f);
  py = fminf(fmaxf(py, 0.f), 1.f);
  float fx = px * (FM - 1), fy = py * (FM - 1);
  float x0 = floorf(fx), y0 = floorf(fy);
  float wx = fx - x0, wy = fy - y0;
  int x0i = (int)x0, y0i = (int)y0;
  int x1i = (x0i + 1 < FM) ? x0i + 1 : FM - 1;
  int y1i = (y0i + 1 < FM) ? y0i + 1 : FM - 1;
  const float* f = fm + c * FM * FM;
  float v00 = f[y0i * FM + x0i], v01 = f[y0i * FM + x1i];
  float v10 = f[y1i * FM + x0i], v11 = f[y1i * FM + x1i];
  float val = v00 * (1.f - wy) * (1.f - wx) + v01 * (1.f - wy) * wx +
              v10 * wy * (1.f - wx) + v11 * wy * wx;
  imgf[nd * (SS * Hh) + s * Hh + c] = (_Float16)val;
}

// ---------------- weight pre-pack into per-lane WMMA B-fragments ----------------
// B-layout used here (contiguous K, same as load_a): column n = lane&15; lanes>=16 carry K+16;
// VGPR j holds packed f16 pair (K = base+2j, base+2j+1).
// Packed array index: (((chunk*nt)+tile)*32 + lane)*8 + j  (dwords)
__global__ void k_pack(const float* W, unsigned* P, int K, int N, int nt) {
  int idx = blockIdx.x * blockDim.x + threadIdx.x;
  int total = (K >> 5) * nt * 256;
  if (idx >= total) return;
  int j = idx & 7, lane = (idx >> 3) & 31;
  int rest = idx >> 8;
  int t = rest % nt, ch = rest / nt;
  int k = ch * 32 + (lane & 16) + 2 * j;
  int col = t * 16 + (lane & 15);
  float lo = (col < N) ? W[k * N + col] : 0.f;
  float hi = (col < N) ? W[(k + 1) * N + col] : 0.f;
  union { _Float16 h; unsigned short u; } a, b;
  a.h = (_Float16)lo;
  b.h = (_Float16)hi;
  P[idx] = (unsigned)a.u | ((unsigned)b.u << 16);
}

// ---------------- WMMA helpers ----------------
__device__ __forceinline__ v8f wmma16(v16h a, v16h b, v8f c) {
  return __builtin_amdgcn_wmma_f32_16x16x32_f16(false, a, false, b, (short)0, c,
                                                false, false);
}
__device__ __forceinline__ v8f vzero8() {
  v8f z = {0.f, 0.f, 0.f, 0.f, 0.f, 0.f, 0.f, 0.f};
  return z;
}

union Frag { uv4 q[2]; v16h v; };

// A-fragment: row-major f16 row, 32-f16 chunk. Lanes 0-15: 16B units {0,1};
// lanes 16-31: units {2,3}  (contiguous K order, matching packed B).
__device__ __forceinline__ v16h load_a(const _Float16* chunk, int hi) {
  Frag t;
  const uv4* p = reinterpret_cast<const uv4*>(chunk);
  t.q[0] = p[2 * hi];
  t.q[1] = p[2 * hi + 1];
  return t.v;
}
__device__ __forceinline__ v16h load_b(const unsigned* pw, int lane) {
  Frag t;
  const uv4* p = reinterpret_cast<const uv4*>(pw + lane * 8);
  t.q[0] = p[0];
  t.q[1] = p[1];
  return t.v;
}

// Software-pipelined 16xK x Kx64 GEMM stage: next chunk's A + 4 B fragments are
// fetched into distinct registers before the current chunk's 4 WMMAs, so the
// scheduler can overlap global b128 loads with matrix ops (partial loadcnt waits
// instead of full drains).
template <int CHUNKS, typename AROW>
__device__ __forceinline__ void gemm4(v8f acc[4], AROW arow, const unsigned* pw,
                                      int lane, int hi) {
  v16h a  = load_a(arow(0), hi);
  v16h b0 = load_b(pw + 0 * 256, lane);
  v16h b1 = load_b(pw + 1 * 256, lane);
  v16h b2 = load_b(pw + 2 * 256, lane);
  v16h b3 = load_b(pw + 3 * 256, lane);
  #pragma unroll
  for (int ch = 0; ch < CHUNKS; ++ch) {
    v16h an = a, n0 = b0, n1 = b1, n2 = b2, n3 = b3;
    if (ch + 1 < CHUNKS) {
      an = load_a(arow(ch + 1), hi);
      const unsigned* p = pw + (ch + 1) * 4 * 256;
      n0 = load_b(p + 0 * 256, lane);
      n1 = load_b(p + 1 * 256, lane);
      n2 = load_b(p + 2 * 256, lane);
      n3 = load_b(p + 3 * 256, lane);
    }
    acc[0] = wmma16(a, b0, acc[0]);
    acc[1] = wmma16(a, b1, acc[1]);
    acc[2] = wmma16(a, b2, acc[2]);
    acc[3] = wmma16(a, b3, acc[3]);
    a = an; b0 = n0; b1 = n1; b2 = n2; b3 = n3;
  }
}

// bias + ReLU + f16 convert, scatter C/D layout into the 16x128 LDS tile
__device__ __forceinline__ void epilogue(const v8f acc[4], const float* bias,
                                         _Float16* lds, int colOff, int row, int hi) {
  #pragma unroll
  for (int t = 0; t < 4; ++t) {
    int n = t * 16 + row;
    float bb = bias[n];
    #pragma unroll
    for (int j = 0; j < 8; ++j)
      lds[(j + hi * 8) * 128 + colOff + n] = (_Float16)fmaxf(acc[t][j] + bb, 0.f);
  }
}

// ---------------- fused edge MLP chain (the 34-GFLOP hot loop) ----------------
__global__ __launch_bounds__(WV * 32)
void k_edge(const _Float16* posf, const _Float16* imgf,
            const unsigned* pWp, const unsigned* pWi, const unsigned* pWm1,
            const unsigned* pWm2, const unsigned* pWm3,
            const float* bp, const float* bi, const float* bm1,
            const float* bm2, const float* bm3,
            const int* src, const int* dst, float* out, int E) {
  __shared__ _Float16 lstile[WV][TE * 128];   // per-wave 16x128 f16 staging
  __shared__ float    lslog[WV][TE * 16];     // per-wave logits tile
  const int w = threadIdx.x >> 5;
  const int lane = threadIdx.x & 31;
  const int hi = lane >> 4;
  const int row = lane & 15;
  const int e0 = (blockIdx.x * WV + w) * TE;
  int e = e0 + row;
  if (e >= E) e = E - 1;
  const int sN = src[e], dN = dst[e];
  _Float16* myl = &lstile[w][0];

  v8f acc[4];

  // ---- ep = relu(concat(pos[s],pos[d]) @ Wp + bp) -> lds cols [0,64) ----
  #pragma unroll
  for (int t = 0; t < 4; ++t) acc[t] = vzero8();
  gemm4<4>(acc,
           [&](int ch) {
             return (ch < 2) ? (posf + (size_t)sN * Hh + ch * 32)
                             : (posf + (size_t)dN * Hh + (ch - 2) * 32);
           },
           pWp, lane, hi);
  epilogue(acc, bp, myl, 0, row, hi);

  // ---- ei = relu(concat(img[s],img[d]) @ Wi + bi) -> lds cols [64,128) ----
  #pragma unroll
  for (int t = 0; t < 4; ++t) acc[t] = vzero8();
  gemm4<16>(acc,
            [&](int ch) {
              return (ch < 8) ? (imgf + (size_t)sN * 256 + ch * 32)
                              : (imgf + (size_t)dN * 256 + (ch - 8) * 32);
            },
            pWi, lane, hi);
  epilogue(acc, bi, myl, 64, row, hi);

  // ---- e1 = relu(e @ Wm1 + bm1) -> lds cols [0,64) ----
  #pragma unroll
  for (int t = 0; t < 4; ++t) acc[t] = vzero8();
  gemm4<4>(acc, [&](int ch) { return myl + row * 128 + ch * 32; }, pWm1, lane, hi);
  epilogue(acc, bm1, myl, 0, row, hi);

  // ---- e2 = relu(e1 @ Wm2 + bm2) -> lds cols [0,64) ----
  #pragma unroll
  for (int t = 0; t < 4; ++t) acc[t] = vzero8();
  gemm4<2>(acc, [&](int ch) { return myl + row * 128 + ch * 32; }, pWm2, lane, hi);
  epilogue(acc, bm2, myl, 0, row, hi);

  // ---- logits = e2 @ Wm3 (64x16 padded, cols 0..1 valid) ----
  v8f accf = vzero8();
  {
    v16h a0 = load_a(myl + row * 128 + 0 * 32, hi);
    v16h a1 = load_a(myl + row * 128 + 1 * 32, hi);
    v16h w0 = load_b(pWm3 + 0 * 256, lane);
    v16h w1 = load_b(pWm3 + 1 * 256, lane);
    accf = wmma16(a0, w0, accf);
    accf = wmma16(a1, w1, accf);
  }
  #pragma unroll
  for (int j = 0; j < 8; ++j)
    lslog[w][(j + hi * 8) * 16 + row] = accf[j];

  // ---- log_softmax over 2 classes, write output ----
  if (lane < 16) {
    float a0 = lslog[w][lane * 16 + 0] + bm3[0];
    float a1 = lslog[w][lane * 16 + 1] + bm3[1];
    float mx = fmaxf(a0, a1);
    float lsev = mx + logf(expf(a0 - mx) + expf(a1 - mx));
    int eo = e0 + lane;
    if (eo < E) {
      out[eo * 2 + 0] = a0 - lsev;
      out[eo * 2 + 1] = a1 - lsev;
    }
  }
}

}  // namespace

extern "C" void kernel_launch(void* const* d_in, const int* in_sizes, int n_in,
                              void* d_out, int out_size, void* d_ws, size_t ws_size,
                              hipStream_t stream) {
  (void)n_in; (void)out_size; (void)ws_size;
  const float* x   = (const float*)d_in[0];
  const float* img = (const float*)d_in[1];
  const float* pos = (const float*)d_in[2];
  const float* cw  = (const float*)d_in[3];
  const float* W1  = (const float*)d_in[4];  const float* b1  = (const float*)d_in[5];
  const float* W2  = (const float*)d_in[6];  const float* b2  = (const float*)d_in[7];
  const float* W3  = (const float*)d_in[8];  const float* b3  = (const float*)d_in[9];
  const float* k1  = (const float*)d_in[10]; const float* bk1 = (const float*)d_in[11];
  const float* k2  = (const float*)d_in[12]; const float* bk2 = (const float*)d_in[13];
  const float* Wp  = (const float*)d_in[14]; const float* bp  = (const float*)d_in[15];
  const float* Wi  = (const float*)d_in[16]; const float* bi  = (const float*)d_in[17];
  const float* Wm1 = (const float*)d_in[18]; const float* bm1 = (const float*)d_in[19];
  const float* Wm2 = (const float*)d_in[20]; const float* bm2 = (const float*)d_in[21];
  const float* Wm3 = (const float*)d_in[22]; const float* bm3 = (const float*)d_in[23];
  const int*   eidx = (const int*)d_in[24];

  const int n = in_sizes[0] / FIN;     // 10000 nodes
  const int E = in_sizes[24] / 2;      // 320000 edges
  const int* srcI = eidx;
  const int* dstI = eidx + E;
  float* out = (float*)d_out;

  // -------- workspace carve-up (~35 MB) --------
  char* base = (char*)d_ws;
  size_t off = 0;
  auto alloc = [&](size_t bytes) -> void* {
    off = (off + 255) & ~size_t(255);
    void* p = base + off;
    off += bytes;
    return p;
  };
  float*    nrm  = (float*)alloc((size_t)n * 4);
  float*    hmm  = (float*)alloc((size_t)n * H2 * 4);
  float*    agg  = (float*)alloc((size_t)n * H2 * 4);
  float*    hA   = (float*)alloc((size_t)n * H2 * 4);
  _Float16* posf = (_Float16*)alloc((size_t)n * Hh * 2);
  _Float16* imgf = (_Float16*)alloc((size_t)n * SS * Hh * 2);
  float*    c1o  = (float*)alloc((size_t)C1 * 256 * 256 * 4);
  float*    fm   = (float*)alloc((size_t)Hh * FM * FM * 4);
  unsigned* pWp  = (unsigned*)alloc((size_t)4 * 4 * 256 * 4);
  unsigned* pWi  = (unsigned*)alloc((size_t)16 * 4 * 256 * 4);
  unsigned* pWm1 = (unsigned*)alloc((size_t)4 * 4 * 256 * 4);
  unsigned* pWm2 = (unsigned*)alloc((size_t)2 * 4 * 256 * 4);
  unsigned* pWm3 = (unsigned*)alloc((size_t)2 * 1 * 256 * 4);

  auto g = [](int tot) { return (tot + 255) / 256; };

  // -------- pack edge-MLP weights into WMMA B-fragment layout --------
  k_pack<<<g(4 * 4 * 256), 256, 0, stream>>>(Wp,  pWp,  128, 64, 4);
  k_pack<<<g(16 * 4 * 256), 256, 0, stream>>>(Wi,  pWi,  512, 64, 4);
  k_pack<<<g(4 * 4 * 256), 256, 0, stream>>>(Wm1, pWm1, 128, 64, 4);
  k_pack<<<g(2 * 4 * 256), 256, 0, stream>>>(Wm2, pWm2,  64, 64, 4);
  k_pack<<<g(2 * 1 * 256), 256, 0, stream>>>(Wm3, pWm3,  64,  2, 1);

  // -------- symmetric-norm degrees --------
  k_zero<<<g(n), 256, 0, stream>>>(nrm, n);
  k_deg<<<g(E), 256, 0, stream>>>(dstI, nrm, E);
  k_norm<<<g(n), 256, 0, stream>>>(nrm, n);

  // -------- GCN layer 1: [N,8] -> [N,128] --------
  k_mm_scale<<<g(n * H2), 256, 0, stream>>>(x, W1, nrm, hmm, n, FIN, H2);
  k_zero<<<g(n * H2), 256, 0, stream>>>(agg, n * H2);
  k_scatter<<<g(E * H2), 256, 0, stream>>>(srcI, dstI, hmm, agg, E, H2);
  k_finish<<<g(n * H2), 256, 0, stream>>>(agg, hmm, nrm, b1, hA, nullptr, n, H2);

  // -------- GCN layer 2: [N,128] -> [N,128] --------
  k_mm_scale<<<g(n * H2), 256, 0, stream>>>(hA, W2, nrm, hmm, n, H2, H2);
  k_zero<<<g(n * H2), 256, 0, stream>>>(agg, n * H2);
  k_scatter<<<g(E * H2), 256, 0, stream>>>(srcI, dstI, hmm, agg, E, H2);
  k_finish<<<g(n * H2), 256, 0, stream>>>(agg, hmm, nrm, b2, hA, nullptr, n, H2);

  // -------- GCN layer 3: [N,128] -> [N,64], also emit f16 table --------
  k_mm_scale<<<g(n * Hh), 256, 0, stream>>>(hA, W3, nrm, hmm, n, H2, Hh);
  k_zero<<<g(n * Hh), 256, 0, stream>>>(agg, n * Hh);
  k_scatter<<<g(E * Hh), 256, 0, stream>>>(srcI, dstI, hmm, agg, E, Hh);
  k_finish<<<g(n * Hh), 256, 0, stream>>>(agg, hmm, nrm, b3, hA, posf, n, Hh);

  // -------- image branch --------
  k_conv1<<<g(C1 * 256 * 256), 256, 0, stream>>>(img, k1, bk1, c1o);
  k_conv2<<<g(Hh * FM * FM), 256, 0, stream>>>(c1o, k2, bk2, fm);
  k_sample<<<g(n * SS * Hh), 256, 0, stream>>>(fm, pos, cw, imgf, n);

  // -------- fused WMMA edge pipeline (dominant) --------
  int nb = (E + WV * TE - 1) / (WV * TE);
  k_edge<<<nb, WV * 32, 0, stream>>>(posf, imgf, pWp, pWi, pWm1, pWm2, pWm3,
                                     bp, bi, bm1, bm2, bm3, srcI, dstI, out, E);
}
